// MultiHeadAttention_8400956031164
// MI455X (gfx1250) — matrix operand order repaired and checked by reading, not compiled, not run on hardware
//
#include <hip/hip_runtime.h>
#include <hip/hip_bf16.h>

// Problem constants (from reference)
#define B_    2
#define S_    2048
#define D_    2048
#define H_    16
#define HD_   128
#define WIN_  1024
#define NROW  (B_*S_)          // 4096 rows of X
#define NCOL  2048             // H_*HD_ == D_
#define SCALE_ 0.08838834764831845f   // 128^-0.5

typedef __attribute__((ext_vector_type(4)))  __bf16 v4bf;
typedef __attribute__((ext_vector_type(8)))  __bf16 v8bf;
typedef __attribute__((ext_vector_type(16))) __bf16 v16bf;
typedef __attribute__((ext_vector_type(8)))  float  v8f;

// ---------- bf16 helpers ----------
__device__ __forceinline__ __bf16 f2bfv(float f) {
  unsigned int u = __float_as_uint(f);
  u += 0x7FFFu + ((u >> 16) & 1u);          // round-to-nearest-even
  unsigned short s = (unsigned short)(u >> 16);
  union { unsigned short s; __bf16 b; } c; c.s = s; return c.b;
}
__device__ __forceinline__ v8bf ld8(const __bf16* p) {     // 16B-aligned 8x bf16
  return *(const v8bf*)p;
}
__device__ __forceinline__ v16bf frag_cat(v8bf lo, v8bf hi) {
  return __builtin_shufflevector(lo, hi, 0,1,2,3,4,5,6,7,8,9,10,11,12,13,14,15);
}

// ---------- CDNA5 async global->LDS copy (16B per lane, ASYNCcnt tracked) ----------
// The clang builtin exists but takes pointers in builtin-only language address
// spaces; inline asm sidesteps the type-system friction. ISA (15.18.3 op 98):
//   GLOBAL_LOAD_ASYNC_TO_LDS_B128 vdst(=LDS byte addr), vaddr(64b global), off
// Flat->LDS mapping uses addr[31:0], so the low 32 bits of the generic LDS
// pointer are the LDS byte address.
__device__ __forceinline__ void async_cp16(const __bf16* g, __bf16* l) {
  asm volatile("global_load_async_to_lds_b128 %0, %1, off"
               :
               : "v"((unsigned)(unsigned long long)l),
                 "v"((unsigned long long)g)
               : "memory");
}
__device__ __forceinline__ void async_wait0() {
#if __has_builtin(__builtin_amdgcn_s_wait_asynccnt)
  __builtin_amdgcn_s_wait_asynccnt(0);
#else
  asm volatile("s_wait_asynccnt 0x0" ::: "memory");
#endif
}

// ---------- kernel 1: fp32 -> bf16 convert (x4 vectorized) ----------
__global__ __launch_bounds__(256) void cvt_bf16_kernel(
    const float* __restrict__ src, __bf16* __restrict__ dst, int n4) {
  int i = blockIdx.x * 256 + threadIdx.x;
  if (i < n4) {
    float4 f = ((const float4*)src)[i];
    v4bf o;
    o[0] = f2bfv(f.x); o[1] = f2bfv(f.y); o[2] = f2bfv(f.z); o[3] = f2bfv(f.w);
    ((v4bf*)dst)[i] = o;
  }
}

// ---------- shared GEMM mainloop: C(128x128) += A(128xK)*B(Kx128), bf16 in, f32 acc ----------
// 256 threads = 8 waves (4m x 2n); each wave: 2x4 WMMA 16x16 tiles.
// A staged [M=128][K=32] row-major via ASYNC copy; B staged TRANSPOSED [N=128][K=32]
// (sync, VGPR bounce) so all fragment loads are contiguous 16B ds_load_b128 pairs.
#define LDA_  40    // 32 + 8 pad (mult of 8 -> aligned b128)
#define LDBT_ 40

__device__ __forceinline__ void gemm_mainloop(
    const __bf16* __restrict__ A,    // [NROW][2048] bf16 row-major
    const __bf16* __restrict__ Bm,   // [2048][2048] bf16 row-major (K x N)
    __bf16* As, __bf16* Bst, v8f acc[2][4]) {
  const int t = threadIdx.x, lane = t & 31, w = t >> 5;
  const int wm = w & 3, wn = w >> 2;
  const int m0 = blockIdx.x * 128, n0 = blockIdx.y * 128;
  const int g = lane >> 4, nl = lane & 15;

  for (int k0 = 0; k0 < D_; k0 += 32) {
    // stage A 128x32: async global->LDS, overlapped with the B transpose below
#pragma unroll
    for (int i = 0; i < 2; ++i) {
      int idx = i * 256 + t;                  // 0..511
      int r = idx >> 2, c8 = idx & 3;
      async_cp16(&A[(size_t)(m0 + r) * D_ + k0 + c8 * 8], &As[r * LDA_ + c8 * 8]);
    }
    // stage B 32x128 transposed: vector global read + 8 scalar LDS writes
#pragma unroll
    for (int i = 0; i < 2; ++i) {
      int idx = i * 256 + t;                  // 0..511
      int r = idx >> 4, c8 = idx & 15;        // k-row, n-chunk
      v8bf vb = ld8(&Bm[(size_t)(k0 + r) * NCOL + n0 + c8 * 8]);
#pragma unroll
      for (int q = 0; q < 8; ++q) Bst[(c8 * 8 + q) * LDBT_ + r] = vb[q];
    }
    if (k0 + 32 < D_) {   // -> global_prefetch_b8
      __builtin_prefetch(&A[(size_t)(m0 + (t >> 1)) * D_ + k0 + 32], 0, 0);
      __builtin_prefetch(&Bm[(size_t)(k0 + 32 + (t >> 3)) * NCOL + n0], 0, 0);
    }
    async_wait0();
    __syncthreads();

    // A fragments: two b128 loads per frag
    v16bf af[2];
#pragma unroll
    for (int i = 0; i < 2; ++i) {
      const __bf16* tp = &As[(wm * 32 + i * 16 + nl) * LDA_ + 8 * g];
      af[i] = frag_cat(ld8(tp), ld8(tp + 16));
    }
    // B fragments from transposed tile: two b128 loads per frag
#pragma unroll
    for (int j = 0; j < 4; ++j) {
      const __bf16* tp = &Bst[(wn * 64 + j * 16 + nl) * LDBT_ + 8 * g];
      v16bf bf = frag_cat(ld8(tp), ld8(tp + 16));
#pragma unroll
      for (int i = 0; i < 2; ++i)
        acc[i][j] = __builtin_amdgcn_wmma_f32_16x16x32_bf16(
            false, af[i], false, bf, (short)0, acc[i][j], false, false);
    }
    __syncthreads();
  }
}

// ---------- kernel 2: QKV projections ----------
// z=0 -> Qf (f32), z=1 -> Kf (f32), z=2 -> Vb (bf16, [B][H][S][HD]; N-tile==head)
__global__ __launch_bounds__(256) void gemm_qkv_kernel(
    const __bf16* __restrict__ Xb,
    const __bf16* __restrict__ Wqb,
    const __bf16* __restrict__ Wkb,
    const __bf16* __restrict__ Wvb,
    float* __restrict__ Qf, float* __restrict__ Kf,
    __bf16* __restrict__ Vb) {
  __shared__ alignas(16) __bf16 As[128 * LDA_];
  __shared__ alignas(16) __bf16 Bst[128 * LDBT_];
  v8f acc[2][4] = {};
  const __bf16* Wb = (blockIdx.z == 0) ? Wqb : (blockIdx.z == 1) ? Wkb : Wvb;
  gemm_mainloop(Xb, Wb, As, Bst, acc);

  const int t = threadIdx.x, lane = t & 31, w = t >> 5;
  const int wm = w & 3, wn = w >> 2;
  const int m0 = blockIdx.x * 128, n0 = blockIdx.y * 128;
  const int g = lane >> 4, nl = lane & 15;

  if (blockIdx.z < 2) {
    float* Cf = (blockIdx.z == 0) ? Qf : Kf;
#pragma unroll
    for (int i = 0; i < 2; ++i)
#pragma unroll
      for (int j = 0; j < 4; ++j)
#pragma unroll
        for (int r = 0; r < 8; ++r) {
          int row = m0 + wm * 32 + i * 16 + r + 8 * g;
          int col = n0 + wn * 64 + j * 16 + nl;
          Cf[(size_t)row * NCOL + col] = acc[i][j][r];
        }
  } else {
    int h = blockIdx.y;
#pragma unroll
    for (int i = 0; i < 2; ++i)
#pragma unroll
      for (int j = 0; j < 4; ++j)
#pragma unroll
        for (int r = 0; r < 8; ++r) {
          int row = m0 + wm * 32 + i * 16 + r + 8 * g;   // b*S + s
          int b = row >> 11, s = row & (S_ - 1);
          int d = wn * 64 + j * 16 + nl;
          Vb[((size_t)(b * H_ + h) * S_ + s) * HD_ + d] = f2bfv(acc[i][j][r]);
        }
  }
}

// ---------- kernel 3: RMSNorm + RoPE for Q/K, relayout to [B][H][S][HD] bf16 ----------
__global__ __launch_bounds__(128) void norm_rope_kernel(
    const float* __restrict__ Qf, const float* __restrict__ Kf,
    const float* __restrict__ cosp, const float* __restrict__ sinp,
    const float* __restrict__ qsc, const float* __restrict__ ksc,
    __bf16* __restrict__ Qb, __bf16* __restrict__ Kb) {
  const int d = threadIdx.x;           // 0..127
  const int x = blockIdx.x;            // b*S + s
  const int h = blockIdx.y;
  const int z = blockIdx.z;            // 0=Q, 1=K
  const float* src = z ? Kf : Qf;
  const float* sc  = z ? ksc : qsc;
  __bf16* dst = z ? Kb : Qb;

  float v = src[(size_t)x * NCOL + h * HD_ + d];
  __shared__ float sred[128];
  __shared__ float shn[128];
  sred[d] = v * v;
  __syncthreads();
#pragma unroll
  for (int off = 64; off > 0; off >>= 1) {
    if (d < off) sred[d] += sred[d + off];
    __syncthreads();
  }
  float rms = rsqrtf(sred[0] * (1.0f / 128.0f) + 1e-6f);
  float xn = sc[d] * v * rms;
  shn[d] = xn;
  __syncthreads();
  float rot = (d < 64) ? -shn[d + 64] : shn[d - 64];
  float c  = cosp[(size_t)x * HD_ + d];
  float sn = sinp[(size_t)x * HD_ + d];
  int b = x >> 11, s = x & (S_ - 1);
  dst[((size_t)(b * H_ + h) * S_ + s) * HD_ + d] = f2bfv(xn * c + rot * sn);
}

// ---------- kernel 4: sliding-window flash attention ----------
// grid (S/128, H, B), 256 threads = 8 waves; wave w owns query rows [q0+16w, q0+16w+15].
// K staged [key][d] via ASYNC copy (contraction d contiguous); V staged TRANSPOSED
// [d][key] (sync, contraction key contiguous) -> all WMMA frag loads are b128 pairs.
#define LDK_  136   // 128 + 8
#define LDVT_ 72    // 64 + 8
#define LDP_  72    // 64 + 8
__global__ __launch_bounds__(256) void attn_kernel(
    const __bf16* __restrict__ Qb,
    const __bf16* __restrict__ Kb,
    const __bf16* __restrict__ Vb,
    __bf16* __restrict__ Ab) {
  __shared__ alignas(16) __bf16 Ks[64 * LDK_];
  __shared__ alignas(16) __bf16 Vst[128 * LDVT_];
  __shared__ alignas(16) __bf16 Ps[8 * 16 * LDP_];

  const int t = threadIdx.x, lane = t & 31, w = t >> 5;
  const int g = lane >> 4, nl = lane & 15;
  const int q0 = blockIdx.x * 128;
  const int h = blockIdx.y, b = blockIdx.z;
  const size_t hoff = (size_t)(b * H_ + h) * S_ * HD_;
  const __bf16* Qh = Qb + hoff;
  const __bf16* Kh = Kb + hoff;
  const __bf16* Vh = Vb + hoff;
  const int qw = q0 + w * 16;

  // preload wave's Q rows as four 16x32 A-fragments (two b128 global loads each)
  v16bf qf[4];
#pragma unroll
  for (int c = 0; c < 4; ++c) {
    const __bf16* qp = &Qh[(size_t)(qw + nl) * HD_ + c * 32 + 8 * g];
    qf[c] = frag_cat(ld8(qp), ld8(qp + 16));
  }

  float mi[8], li[8];
  v8f oacc[8] = {};
#pragma unroll
  for (int r = 0; r < 8; ++r) { mi[r] = -1e30f; li[r] = 0.0f; }

  int kb0 = q0 - WIN_; if (kb0 < 0) kb0 = 0;
  for (int kb = kb0; kb <= q0 + 64; kb += 64) {
    // stage K 64x128 (async copy) and V transposed 128x64 (sync bounce, overlapped)
#pragma unroll
    for (int i = 0; i < 4; ++i) {
      int idx = i * 256 + t;                  // 0..1023
      int r = idx >> 4, c8 = idx & 15;
      async_cp16(&Kh[(size_t)(kb + r) * HD_ + c8 * 8], &Ks[r * LDK_ + c8 * 8]);
    }
#pragma unroll
    for (int i = 0; i < 4; ++i) {
      int idx = i * 256 + t;
      int r = idx >> 4, c8 = idx & 15;
      v8bf vv = ld8(&Vh[(size_t)(kb + r) * HD_ + c8 * 8]);
#pragma unroll
      for (int q = 0; q < 8; ++q) Vst[(c8 * 8 + q) * LDVT_ + r] = vv[q];
    }
    async_wait0();
    __syncthreads();

    // S = Q*K^T : 4 N-tiles x 4 K-chains
    v8f sacc[4] = {};
#pragma unroll
    for (int j = 0; j < 4; ++j)
#pragma unroll
      for (int c = 0; c < 4; ++c) {
        const __bf16* tp = &Ks[(j * 16 + nl) * LDK_ + c * 32 + 8 * g];
        v16bf bk = frag_cat(ld8(tp), ld8(tp + 16));
        sacc[j] = __builtin_amdgcn_wmma_f32_16x16x32_bf16(
            false, qf[c], false, bk, (short)0, sacc[j], false, false);
      }

    // online softmax (row stats stay within the 16-lane half that owns the row)
    float pv[4][8];
#pragma unroll
    for (int r = 0; r < 8; ++r) {
      int qg = qw + r + 8 * g;
      float sv[4], mx = -1e30f;
#pragma unroll
      for (int j = 0; j < 4; ++j) {
        int kg = kb + j * 16 + nl;
        float s = sacc[j][r] * SCALE_;
        bool valid = (kg <= qg) && (kg + WIN_ >= qg);
        s = valid ? s : -1e30f;
        sv[j] = s;
        mx = fmaxf(mx, s);
      }
      mx = fmaxf(mx, __shfl_xor(mx, 1, 32));
      mx = fmaxf(mx, __shfl_xor(mx, 2, 32));
      mx = fmaxf(mx, __shfl_xor(mx, 4, 32));
      mx = fmaxf(mx, __shfl_xor(mx, 8, 32));
      float mnew = fmaxf(mi[r], mx);
      float rs = 0.0f;
#pragma unroll
      for (int j = 0; j < 4; ++j) {
        float p = __expf(sv[j] - mnew);
        pv[j][r] = p;
        rs += p;
      }
      rs += __shfl_xor(rs, 1, 32);
      rs += __shfl_xor(rs, 2, 32);
      rs += __shfl_xor(rs, 4, 32);
      rs += __shfl_xor(rs, 8, 32);
      float alpha = __expf(mi[r] - mnew);
      li[r] = li[r] * alpha + rs;
      mi[r] = mnew;
#pragma unroll
      for (int nt = 0; nt < 8; ++nt) oacc[nt][r] = oacc[nt][r] * alpha;
    }

    // D-layout f32 -> A-layout bf16 via wave-private LDS staging
#pragma unroll
    for (int j = 0; j < 4; ++j)
#pragma unroll
      for (int r = 0; r < 8; ++r)
        Ps[(w * 16 + r + 8 * g) * LDP_ + j * 16 + nl] = f2bfv(pv[j][r]);

    // O += P*V : 2 K-chains x 8 HD tiles (all frag loads b128 pairs)
#pragma unroll
    for (int c = 0; c < 2; ++c) {
      const __bf16* pp = &Ps[(w * 16 + nl) * LDP_ + c * 32 + 8 * g];
      v16bf ap = frag_cat(ld8(pp), ld8(pp + 16));
#pragma unroll
      for (int nt = 0; nt < 8; ++nt) {
        const __bf16* tp = &Vst[(nt * 16 + nl) * LDVT_ + c * 32 + 8 * g];
        v16bf bv = frag_cat(ld8(tp), ld8(tp + 16));
        oacc[nt] = __builtin_amdgcn_wmma_f32_16x16x32_bf16(
            false, ap, false, bv, (short)0, oacc[nt], false, false);
      }
    }
    __syncthreads();
  }

  // normalize; write bf16 [b][s][h*128 + d] for the Wo GEMM
#pragma unroll
  for (int r = 0; r < 8; ++r) {
    float inv = 1.0f / li[r];
    int row = qw + r + 8 * g;   // s index
#pragma unroll
    for (int nt = 0; nt < 8; ++nt)
      Ab[((size_t)b * S_ + row) * NCOL + h * HD_ + nt * 16 + nl] =
          f2bfv(oacc[nt][r] * inv);
  }
}

// ---------- kernel 5: output projection (fp32 out) ----------
__global__ __launch_bounds__(256) void gemm_out_kernel(
    const __bf16* __restrict__ Ab,
    const __bf16* __restrict__ Wob,
    float* __restrict__ out) {
  __shared__ alignas(16) __bf16 As[128 * LDA_];
  __shared__ alignas(16) __bf16 Bst[128 * LDBT_];
  v8f acc[2][4] = {};
  gemm_mainloop(Ab, Wob, As, Bst, acc);

  const int t = threadIdx.x, lane = t & 31, w = t >> 5;
  const int wm = w & 3, wn = w >> 2;
  const int m0 = blockIdx.x * 128, n0 = blockIdx.y * 128;
  const int g = lane >> 4, nl = lane & 15;
#pragma unroll
  for (int i = 0; i < 2; ++i)
#pragma unroll
    for (int j = 0; j < 4; ++j)
#pragma unroll
      for (int r = 0; r < 8; ++r) {
        int row = m0 + wm * 32 + i * 16 + r + 8 * g;
        int col = n0 + wn * 64 + j * 16 + nl;
        out[(size_t)row * NCOL + col] = acc[i][j][r];
      }
}

// ---------- host launcher ----------
extern "C" void kernel_launch(void* const* d_in, const int* in_sizes, int n_in,
                              void* d_out, int out_size, void* d_ws, size_t ws_size,
                              hipStream_t stream) {
  const float* hs   = (const float*)d_in[0];
  const float* cosp = (const float*)d_in[1];
  const float* sinp = (const float*)d_in[2];
  const float* Wq   = (const float*)d_in[3];
  const float* Wk   = (const float*)d_in[4];
  const float* Wv   = (const float*)d_in[5];
  const float* Wo   = (const float*)d_in[6];
  const float* qsc  = (const float*)d_in[7];
  const float* ksc  = (const float*)d_in[8];
  float* out = (float*)d_out;

  // workspace carve-out (~176 MB)
  char* p = (char*)d_ws;
  __bf16* Xb  = (__bf16*)p; p += (size_t)NROW * D_ * 2;
  __bf16* Wqb = (__bf16*)p; p += (size_t)D_ * NCOL * 2;
  __bf16* Wkb = (__bf16*)p; p += (size_t)D_ * NCOL * 2;
  __bf16* Wvb = (__bf16*)p; p += (size_t)D_ * NCOL * 2;
  __bf16* Wob = (__bf16*)p; p += (size_t)D_ * NCOL * 2;
  float* Qf = (float*)p; p += (size_t)NROW * NCOL * 4;
  float* Kf = (float*)p; p += (size_t)NROW * NCOL * 4;
  __bf16* Qb = (__bf16*)p; p += (size_t)B_ * H_ * S_ * HD_ * 2;
  __bf16* Kb = (__bf16*)p; p += (size_t)B_ * H_ * S_ * HD_ * 2;
  __bf16* Vb = (__bf16*)p; p += (size_t)B_ * H_ * S_ * HD_ * 2;
  __bf16* Ab = (__bf16*)p; p += (size_t)NROW * NCOL * 2;

  const int nX4 = NROW * D_ / 4;
  const int nW4 = D_ * NCOL / 4;
  cvt_bf16_kernel<<<(nX4 + 255) / 256, 256, 0, stream>>>(hs, Xb, nX4);
  cvt_bf16_kernel<<<(nW4 + 255) / 256, 256, 0, stream>>>(Wq, Wqb, nW4);
  cvt_bf16_kernel<<<(nW4 + 255) / 256, 256, 0, stream>>>(Wk, Wkb, nW4);
  cvt_bf16_kernel<<<(nW4 + 255) / 256, 256, 0, stream>>>(Wv, Wvb, nW4);
  cvt_bf16_kernel<<<(nW4 + 255) / 256, 256, 0, stream>>>(Wo, Wob, nW4);

  gemm_qkv_kernel<<<dim3(NROW / 128, NCOL / 128, 3), 256, 0, stream>>>(
      Xb, Wqb, Wkb, Wvb, Qf, Kf, Vb);

  norm_rope_kernel<<<dim3(NROW, H_, 2), 128, 0, stream>>>(
      Qf, Kf, cosp, sinp, qsc, ksc, Qb, Kb);

  attn_kernel<<<dim3(S_ / 128, H_, B_), 256, 0, stream>>>(Qb, Kb, Vb, Ab);

  gemm_out_kernel<<<dim3(NROW / 128, NCOL / 128, 1), 256, 0, stream>>>(Ab, Wob, out);
}
